// KMeansQuantizer_31842887533189
// MI455X (gfx1250) — matrix operand order repaired and checked by reading, not compiled, not run on hardware
//
#include <hip/hip_runtime.h>

// ---------------------------------------------------------------------------
// KMeans predict: argmin_k ( ||c_k||^2 - 2 * x . c_k )
// N=131072 tokens, D=768 dims, K=1024 clusters.
// fp16 WMMA (V_WMMA_F32_16X16X32_F16, f32 accum) GEMM with fused argmin.
//
// Round-5 tuning: add amdgpu_waves_per_eu(1) so the pre-RA scheduler's
// pressure target stops sinking the B-fragment loads onto their uses
// (round-4 asm: single recycled buffer + s_wait_loadcnt 0x0 per WMMA pair).
// Occupancy is LDS-limited (3 WGs / 6 waves per WGP at ~104.5 KB), so
// relaxing the RA occupancy floor is free; unroll-2 on the K-chunk loop
// keeps live state < 256 VGPRs (no s_set_vgpr_msb on the hot path).
// ---------------------------------------------------------------------------

#define N_TOK      131072
#define DIMS       768          // K-dim of GEMM, 24 chunks of 32
#define NCLUST     1024         // 8 groups of 128 clusters
#define TOK_PER_WG 64           // 2 waves x 32 tokens
#define NTHREADS   64
#define XS_BYTES   (TOK_PER_WG * (DIMS / 8) * 16)       // 64 rows * 96 x 16B = 96 KB
#define BEST_BYTES (NTHREADS * 17 * 4)                  // padded stride 17 dwords
#define SMEM_BYTES (XS_BYTES + 2 * BEST_BYTES)          // ~104.5 KB

typedef _Float16 v16h __attribute__((ext_vector_type(16)));
typedef float    v8f  __attribute__((ext_vector_type(8)));
typedef float    v4f  __attribute__((ext_vector_type(4)));

union FragH { v16h h; v4f f4[2]; };

// ---------------------------------------------------------------------------
// Prep: centroids fp32 -> fp16 (row-major [K][D]) + exact fp32 ||c||^2.
// ---------------------------------------------------------------------------
__global__ __launch_bounds__(256) void kmeans_prep(
    const float* __restrict__ cen, _Float16* __restrict__ cw,
    float* __restrict__ csq) {
  const int k = blockIdx.x;
  const int t = threadIdx.x;
  const float* row = cen + (size_t)k * DIMS;
  _Float16* orow   = cw  + (size_t)k * DIMS;
  float s = 0.f;
  for (int i = t; i < DIMS; i += 256) {
    float v = row[i];
    orow[i] = (_Float16)v;
    s = fmaf(v, v, s);
  }
  __shared__ float red[256];
  red[t] = s;
  __syncthreads();
  for (int off = 128; off > 0; off >>= 1) {
    if (t < off) red[t] += red[t + off];
    __syncthreads();
  }
  if (t == 0) csq[k] = red[0];
}

// ---------------------------------------------------------------------------
// Main kernel.
// ---------------------------------------------------------------------------
__global__ __launch_bounds__(NTHREADS)
__attribute__((amdgpu_waves_per_eu(1))) void kmeans_argmin(
    const float* __restrict__ x, const _Float16* __restrict__ cw,
    const float* __restrict__ csq, int* __restrict__ out) {
  extern __shared__ char smem_raw[];
  v4f*   xs16  = reinterpret_cast<v4f*>(smem_raw);  // [row][96] 16B chunks, swizzled
  float* bestL = reinterpret_cast<float*>(smem_raw + XS_BYTES);
  int*   bidxL = reinterpret_cast<int*>(smem_raw + XS_BYTES + BEST_BYTES);

  const int tid    = threadIdx.x;
  const int lane   = tid & 31;
  const int wid    = tid >> 5;
  const int token0 = blockIdx.x * TOK_PER_WG;
  const int bslot  = tid * 17;             // padded: lane addresses sweep 64 banks

  // ---- Phase 1: x tile fp32 -> fp16 into XOR-swizzled LDS.
  const v4f* xg = reinterpret_cast<const v4f*>(x + (size_t)token0 * DIMS);
  for (int i = tid; i < TOK_PER_WG * (DIMS / 8); i += NTHREADS) {
    const int row = i / 96;
    const int c16 = i - row * 96;
    v4f p0 = xg[row * 192 + c16 * 2 + 0];
    v4f p1 = xg[row * 192 + c16 * 2 + 1];
    union { v4f f; _Float16 h[8]; } u;
    u.h[0] = (_Float16)p0[0]; u.h[1] = (_Float16)p0[1];
    u.h[2] = (_Float16)p0[2]; u.h[3] = (_Float16)p0[3];
    u.h[4] = (_Float16)p1[0]; u.h[5] = (_Float16)p1[1];
    u.h[6] = (_Float16)p1[2]; u.h[7] = (_Float16)p1[3];
    xs16[row * 96 + (c16 ^ (row & 15))] = u.f;
  }
#pragma unroll
  for (int s = 0; s < 16; ++s) { bestL[bslot + s] = 3.4e38f; bidxL[bslot + s] = 0; }
  __syncthreads();

  // ---- Phase 2: WMMA GEMM + fused argmin.
  const int col  = lane & 15;   // C-matrix column slot (cluster within 16-tile)
  const int half = lane >> 4;   // K-half selector per ISA A/B fragment layout
  const int rA0  = wid * 32 + col;
  const int rA1  = rA0 + 16;

  const v8f zacc = {0.f, 0.f, 0.f, 0.f, 0.f, 0.f, 0.f, 0.f};

#define LOAD_B(dst, jj)                                                        \
  do {                                                                         \
    const v4f* bp_ =                                                           \
        reinterpret_cast<const v4f*>(cbase + (size_t)(jj) * 16 * DIMS);        \
    (dst).f4[0] = bp_[0];                                                      \
    (dst).f4[1] = bp_[2];                                                      \
  } while (0)

  for (int g = 0; g < 8; ++g) {          // 8 groups of 128 clusters
    v8f acc[2][8];
#pragma unroll
    for (int a = 0; a < 2; ++a)
#pragma unroll
      for (int j = 0; j < 8; ++j) acc[a][j] = zacc;

#pragma unroll 2
    for (int ch = 0; ch < 24; ++ch) {    // K-dim: 24 chunks of 32
      // Prefetch the B stream 2 chunks ahead (one 128B line = 2 chunks of a
      // centroid row); 4 lane-spread prefetches cover all 128 tile rows.
      if (((ch & 1) == 0) && ch < 22) {
        const _Float16* pf =
            cw + (size_t)(g * 128 + lane) * DIMS + (ch + 2) * 32;
#pragma unroll
        for (int q = 0; q < 4; ++q)
          __builtin_prefetch(pf + (size_t)(32 * q) * DIMS, 0, 3);
      }

      const int c16a = ch * 4 + half;
      FragH A0, A1;
      A0.f4[0] = xs16[rA0 * 96 + ((c16a    ) ^ (rA0 & 15))];
      A0.f4[1] = xs16[rA0 * 96 + ((c16a + 2) ^ (rA0 & 15))];
      A1.f4[0] = xs16[rA1 * 96 + ((c16a    ) ^ (rA1 & 15))];
      A1.f4[1] = xs16[rA1 * 96 + ((c16a + 2) ^ (rA1 & 15))];

      // B fragments: lane holds cluster (g*128 + j*16 + col), fp16 dims
      // ch*32 + half*8 + [0..7] and ch*32 + 16 + half*8 + [0..7] (matches A). 3-deep register
      // ring: fragment for step j+2 is in flight while step j computes.
      const _Float16* cbase =
          cw + (size_t)(g * 128 + col) * DIMS + ch * 32 + half * 8;
      FragH Bf[3];
      LOAD_B(Bf[0], 0);
      LOAD_B(Bf[1], 1);
#pragma unroll
      for (int j = 0; j < 8; ++j) {
        if (j < 6) LOAD_B(Bf[(j + 2) % 3], j + 2);
        acc[0][j] = __builtin_amdgcn_wmma_f32_16x16x32_f16(
            false, A0.h, false, Bf[j % 3].h, (short)0, acc[0][j], false, false);
        acc[1][j] = __builtin_amdgcn_wmma_f32_16x16x32_f16(
            false, A1.h, false, Bf[j % 3].h, (short)0, acc[1][j], false, false);
      }
    }

    // Fused scoring: score = ||c||^2 - 2*dot ; running argmin kept in LDS
    // (touched once per group -> frees 32 VGPRs during the GEMM).
#pragma unroll
    for (int j = 0; j < 8; ++j) {
      const int cl   = g * 128 + j * 16 + col;
      const float cs = csq[cl];
#pragma unroll
      for (int v = 0; v < 8; ++v) {
        float s0 = fmaf(-2.f, acc[0][j][v], cs);
        if (s0 < bestL[bslot + v]) { bestL[bslot + v] = s0; bidxL[bslot + v] = cl; }
        float s1 = fmaf(-2.f, acc[1][j][v], cs);
        if (s1 < bestL[bslot + 8 + v]) {
          bestL[bslot + 8 + v] = s1;
          bidxL[bslot + 8 + v] = cl;
        }
      }
    }
  }
#undef LOAD_B

  // ---- Cross-lane argmin finish (C layout: lanes 0-15 hold M=v, lanes
  // 16-31 hold M=v+8; masks 1/2/4/8 reduce within each 16-lane half).
  float best[16];
  int   bidx[16];
#pragma unroll
  for (int s = 0; s < 16; ++s) { best[s] = bestL[bslot + s]; bidx[s] = bidxL[bslot + s]; }

#pragma unroll
  for (int m = 1; m <= 8; m <<= 1) {
#pragma unroll
    for (int s = 0; s < 16; ++s) {
      float os = __shfl_xor(best[s], m, 32);
      int   oi = __shfl_xor(bidx[s], m, 32);
      if (os < best[s] || (os == best[s] && oi < bidx[s])) {
        best[s] = os;
        bidx[s] = oi;
      }
    }
  }

  const int base = token0 + wid * 32;
  if (lane == 0) {
#pragma unroll
    for (int a = 0; a < 2; ++a)
#pragma unroll
      for (int v = 0; v < 8; ++v) out[base + a * 16 + v] = bidx[a * 8 + v];
  } else if (lane == 16) {
#pragma unroll
    for (int a = 0; a < 2; ++a)
#pragma unroll
      for (int v = 0; v < 8; ++v) out[base + a * 16 + 8 + v] = bidx[a * 8 + v];
  }
}

// ---------------------------------------------------------------------------
extern "C" void kernel_launch(void* const* d_in, const int* in_sizes, int n_in,
                              void* d_out, int out_size, void* d_ws, size_t ws_size,
                              hipStream_t stream) {
  const float* x   = (const float*)d_in[0];   // [131072, 768] fp32
  const float* cen = (const float*)d_in[1];   // [1024, 768] fp32

  _Float16* cw = (_Float16*)d_ws;                                   // 1.5 MB fp16 centroids
  float* csq   = (float*)((char*)d_ws + (size_t)NCLUST * DIMS * 2); // 4 KB ||c||^2
  int* out     = (int*)d_out;

  (void)hipFuncSetAttribute((const void*)kmeans_argmin,
                            hipFuncAttributeMaxDynamicSharedMemorySize,
                            SMEM_BYTES);

  kmeans_prep<<<dim3(NCLUST), dim3(256), 0, stream>>>(cen, cw, csq);
  kmeans_argmin<<<dim3(N_TOK / TOK_PER_WG), dim3(NTHREADS), SMEM_BYTES, stream>>>(
      x, cw, csq, out);
}
